// PathIntegration_20375324852565
// MI455X (gfx1250) — matrix operand order repaired and checked by reading, not compiled, not run on hardware
//
#include <hip/hip_runtime.h>
#include <hip/hip_bf16.h>
#include <stdint.h>

typedef __attribute__((ext_vector_type(16))) __bf16 v16bf;
typedef __attribute__((ext_vector_type(8)))  float  v8f;

#define B_    32
#define S_    2048
#define DA    32
#define HID   128
#define OUTD  4096   // 64*64 transition matrix per token
#define NTOK  (B_ * S_)

// ---------------- workspace layout (bytes) ----------------
#define WS_ABF   ((size_t)0)                       // bf16 [65536][32]   = 4 MB
#define WS_W0T   ((size_t)4194304)                 // bf16 [128][32]     = 8 KB
#define WS_W1T   ((size_t)4202496)                 // bf16 [128][128]    = 32 KB
#define WS_W2T   ((size_t)4235264)                 // bf16 [4096][128]   = 1 MB
#define WS_TRANS ((size_t)5283840)                 // bf16 [65536][4096] = 512 MB

// ---------------------------------------------------------------------------
// prep: f32 -> bf16 convert (actions)
__global__ __launch_bounds__(256) void k_cvt(const float* __restrict__ src,
                                             __bf16* __restrict__ dst, int n) {
  int i = blockIdx.x * 256 + threadIdx.x;
  if (i < n) dst[i] = (__bf16)src[i];
}

// prep: f32 [R][C] -> bf16 transposed [C][R]
__global__ __launch_bounds__(256) void k_tcvt(const float* __restrict__ src,
                                              __bf16* __restrict__ dst, int R, int C) {
  int i = blockIdx.x * 256 + threadIdx.x;
  if (i < R * C) {
    int r = i / C, c = i % C;
    dst[c * R + r] = (__bf16)src[i];
  }
}

// ---------------------------------------------------------------------------
// Fused MLP: one wave handles a 16-token tile through all three layers.
// A-fragment (16x32 bf16): lane row m=l&15, half=l>>4; element e = K[16h+e] (same order as B).
// B-fragment (32x16 bf16): lane col n=l&15, half=l>>4; element e = K[16h+e].
// C/D (16x16 f32): VGPR r -> row m = r + 8*half, col n = l&15.
//
// Layers 0/1: rows = tokens, cols = hidden units (epilogue scatters to LDS H).
// Layer 2 is TRANSPOSED: C[o_tile, token] = W2T_tile(A) x H1^T(B) so each
// lane's 8 accumulators are 8 consecutive output columns o of ONE token ->
// a single packed 16-byte store per tile.
__global__ __launch_bounds__(256) void k_mlp(const __bf16* __restrict__ Abf,
                                             const __bf16* __restrict__ W0T,
                                             const __bf16* __restrict__ W1T,
                                             const __bf16* __restrict__ W2T,
                                             const float* __restrict__ b0,
                                             const float* __restrict__ b1,
                                             const float* __restrict__ b2,
                                             __bf16* __restrict__ Trans) {
  __shared__ __bf16 Hs[8][16 * HID];   // 4 KB per wave
  const int lane = threadIdx.x & 31;
  const int wv   = threadIdx.x >> 5;
  const int n    = lane & 15;
  const int half = lane >> 4;
  const int tile = blockIdx.x * 8 + wv;
  const size_t tbase = (size_t)tile * 16;
  __bf16* H = &Hs[wv][0];

  auto load2 = [](const void* p0, const void* p1) -> v16bf {
    union { v16bf v; uint4 q[2]; } u;
    u.q[0] = *(const uint4*)p0;
    u.q[1] = *(const uint4*)p1;
    return u.v;
  };

  // ---- layer 0: A[16,32] @ W0[32,128] ----
  const char* arow = (const char*)Abf + (tbase + (size_t)n) * (DA * 2);
  v16bf a0 = load2(arow + 32 * half, arow + 16 + 32 * half);
  for (int ct = 0; ct < 8; ++ct) {
    const char* brow = (const char*)W0T + (size_t)(ct * 16 + n) * 64 + half * 32;
    v16bf b = load2(brow, brow + 16);
    v8f acc = {0.f, 0.f, 0.f, 0.f, 0.f, 0.f, 0.f, 0.f};
    acc = __builtin_amdgcn_wmma_f32_16x16x32_bf16(false, a0, false, b,
                                                  (short)0, acc, false, false);
    float bias = b0[ct * 16 + n];
#pragma unroll
    for (int r = 0; r < 8; ++r) {
      float v = acc[r] + bias;
      v = v > 0.f ? v : 0.f;
      H[(r + 8 * half) * HID + ct * 16 + n] = (__bf16)v;
    }
  }

  // ---- layer 1: H0[16,128] @ W1[128,128] ----
  v16bf a1[4];
#pragma unroll
  for (int kc = 0; kc < 4; ++kc) {
    const char* hrow = (const char*)H + n * (HID * 2) + kc * 64 + 32 * half;
    a1[kc] = load2(hrow, hrow + 16);
  }
  float h1v[8][8];
#pragma unroll
  for (int ct = 0; ct < 8; ++ct) {
    v8f acc = {0.f, 0.f, 0.f, 0.f, 0.f, 0.f, 0.f, 0.f};
#pragma unroll
    for (int kc = 0; kc < 4; ++kc) {
      const char* brow = (const char*)W1T + (size_t)(ct * 16 + n) * 256 + kc * 64 + half * 32;
      v16bf b = load2(brow, brow + 16);
      acc = __builtin_amdgcn_wmma_f32_16x16x32_bf16(false, a1[kc], false, b,
                                                    (short)0, acc, false, false);
    }
    float bias = b1[ct * 16 + n];
#pragma unroll
    for (int r = 0; r < 8; ++r) {
      float v = acc[r] + bias;
      h1v[ct][r] = v > 0.f ? v : 0.f;
    }
  }
#pragma unroll
  for (int ct = 0; ct < 8; ++ct)
#pragma unroll
    for (int r = 0; r < 8; ++r)
      H[(r + 8 * half) * HID + ct * 16 + n] = (__bf16)h1v[ct][r];

  // ---- layer 2 (transposed): C[o,t] = W2T_tile @ H1^T ----
  // B operand = H1 (loop-invariant, hoisted): lane token n, K contiguous.
  v16bf bh[4];
#pragma unroll
  for (int kc = 0; kc < 4; ++kc) {
    const char* hrow = (const char*)H + n * (HID * 2) + kc * 64 + 32 * half;
    bh[kc] = load2(hrow, hrow + 16);
  }
  const size_t tok = tbase + n;            // this lane's token (C column)
#pragma unroll 4
  for (int ct = 0; ct < 256; ++ct) {
    v8f acc = {0.f, 0.f, 0.f, 0.f, 0.f, 0.f, 0.f, 0.f};
#pragma unroll
    for (int kc = 0; kc < 4; ++kc) {
      // A = W2T rows o = ct*16 + (lane&15); K contiguous.
      const char* arw = (const char*)W2T + (size_t)(ct * 16 + n) * 256 + kc * 64 + 32 * half;
      v16bf a = load2(arw, arw + 16);
      acc = __builtin_amdgcn_wmma_f32_16x16x32_bf16(false, a, false, bh[kc],
                                                    (short)0, acc, false, false);
    }
    // lane holds o = ct*16 + 8*half + r, r=0..7, for token `tok`
    const int obase = ct * 16 + 8 * half;
    const float4 bl0 = *(const float4*)(b2 + obase);
    const float4 bl1 = *(const float4*)(b2 + obase + 4);
    union { uint4 q; __bf16 h[8]; } pk;
    pk.h[0] = (__bf16)(acc[0] + bl0.x);
    pk.h[1] = (__bf16)(acc[1] + bl0.y);
    pk.h[2] = (__bf16)(acc[2] + bl0.z);
    pk.h[3] = (__bf16)(acc[3] + bl0.w);
    pk.h[4] = (__bf16)(acc[4] + bl1.x);
    pk.h[5] = (__bf16)(acc[5] + bl1.y);
    pk.h[6] = (__bf16)(acc[6] + bl1.z);
    pk.h[7] = (__bf16)(acc[7] + bl1.w);
    *(uint4*)(Trans + tok * (size_t)OUTD + obase) = pk.q;
  }
}

// ---------------------------------------------------------------------------
// RNN scan: one block per batch, 512 threads (16 waves). 32-deep async
// global->LDS pipeline (256 KB of the 320 KB WGP LDS) keeps 256 KB in flight
// per sequential chain; 1 async b128 op per wave per step (<=32 outstanding,
// ASYNCcnt is 6-bit).
#define PDEPTH 32
struct RnnSmem {
  __bf16 Tb[PDEPTH][OUTD];  // 32 x 8 KB = 256 KB
  float  partials[512];
  float  vbuf[64];
  float  hbuf[64];
  float  snorm;
};

__global__ __launch_bounds__(512) void k_rnn(const __bf16* __restrict__ Trans,
                                             const float* __restrict__ init_hidden,
                                             float* __restrict__ out) {
  __shared__ RnnSmem sm;
  const int tid = threadIdx.x;
  const int b   = blockIdx.x;
  const int j   = tid & 63;
  const int q   = tid >> 6;      // 0..7

  // h0 = l2norm(init_hidden)
  if (tid < 64) sm.vbuf[tid] = init_hidden[tid];
  __syncthreads();
  if (tid < 32) {
    float a = sm.vbuf[tid], c = sm.vbuf[tid + 32];
    float ss = a * a + c * c;
#pragma unroll
    for (int off = 16; off; off >>= 1) ss += __shfl_xor(ss, off);
    if (tid == 0) sm.snorm = ss;
  }
  __syncthreads();
  if (tid < 64) {
    float nrm = sqrtf(sm.snorm);
    sm.hbuf[tid] = sm.vbuf[tid] / fmaxf(nrm, 1e-12f);
  }
  __syncthreads();

  const uint64_t sbase = (uint64_t)Trans;
  const unsigned ldsTb0 = (unsigned)((char*)&sm.Tb[0][0] - (char*)&sm);
  const unsigned c0 = (unsigned)tid * 16u;   // this thread's 16-byte slice

  auto issue = [&](int s) {
    unsigned goff = (unsigned)(((uint64_t)(b * S_ + s)) << 13) + c0;
    unsigned l0 = ldsTb0 + (unsigned)(s & (PDEPTH - 1)) * 8192u + c0;
    asm volatile("global_load_async_to_lds_b128 %0, %1, %2"
                 :
                 : "v"(l0), "v"(goff), "s"(sbase)
                 : "memory");
  };

  // prologue: fill the pipeline
  for (int s = 0; s < PDEPTH; ++s) issue(s);
  asm volatile("s_wait_asynccnt 31" ::: "memory");  // oldest tile landed (P-1)
  __syncthreads();

  for (int s = 0; s < S_; ++s) {
    const __bf16* T = &sm.Tb[s & (PDEPTH - 1)][0];
    float acc = 0.f;
#pragma unroll
    for (int ii = 0; ii < 8; ++ii) {
      int i = q * 8 + ii;
      acc += sm.hbuf[i] * (float)T[i * 64 + j];
    }
    sm.partials[tid] = acc;
    __syncthreads();                    // tile (s%P) fully consumed
    if (s + PDEPTH < S_) issue(s + PDEPTH);
    if (tid < 64) {
      float v = 0.f;
#pragma unroll
      for (int qq = 0; qq < 8; ++qq) v += sm.partials[tid + 64 * qq];
      sm.vbuf[tid] = fmaxf(v, 0.f);
    }
    __syncthreads();
    if (tid < 32) {
      float a0 = sm.vbuf[tid], a1 = sm.vbuf[tid + 32];
      float ss = a0 * a0 + a1 * a1;
#pragma unroll
      for (int off = 16; off; off >>= 1) ss += __shfl_xor(ss, off);
      if (tid == 0) sm.snorm = ss;
    }
    __syncthreads();
    if (tid < 64) {
      float nrm = sqrtf(sm.snorm);
      float hv = sm.vbuf[tid] / fmaxf(nrm, 1e-12f);
      sm.hbuf[tid] = hv;
      out[((size_t)b * S_ + s) * 64 + tid] = hv;
    }
    asm volatile("s_wait_asynccnt 31" ::: "memory");  // tile (s+1)%P ready
    __syncthreads();
  }
}

// ---------------------------------------------------------------------------
extern "C" void kernel_launch(void* const* d_in, const int* in_sizes, int n_in,
                              void* d_out, int out_size, void* d_ws, size_t ws_size,
                              hipStream_t stream) {
  (void)in_sizes; (void)n_in; (void)out_size; (void)ws_size;
  const float* actions = (const float*)d_in[0];
  const float* w0 = (const float*)d_in[1];
  const float* b0 = (const float*)d_in[2];
  const float* w1 = (const float*)d_in[3];
  const float* b1 = (const float*)d_in[4];
  const float* w2 = (const float*)d_in[5];
  const float* b2 = (const float*)d_in[6];
  const float* ih = (const float*)d_in[7];

  char* ws = (char*)d_ws;
  __bf16* Abf   = (__bf16*)(ws + WS_ABF);
  __bf16* W0T   = (__bf16*)(ws + WS_W0T);
  __bf16* W1T   = (__bf16*)(ws + WS_W1T);
  __bf16* W2T   = (__bf16*)(ws + WS_W2T);
  __bf16* Trans = (__bf16*)(ws + WS_TRANS);

  k_cvt<<<(NTOK * DA + 255) / 256, 256, 0, stream>>>(actions, Abf, NTOK * DA);
  k_tcvt<<<(DA * HID + 255) / 256, 256, 0, stream>>>(w0, W0T, DA, HID);
  k_tcvt<<<(HID * HID + 255) / 256, 256, 0, stream>>>(w1, W1T, HID, HID);
  k_tcvt<<<(HID * OUTD + 255) / 256, 256, 0, stream>>>(w2, W2T, HID, OUTD);
  k_mlp<<<NTOK / (16 * 8), 256, 0, stream>>>(Abf, W0T, W1T, W2T, b0, b1, b2, Trans);
  k_rnn<<<B_, 512, 0, stream>>>(Trans, ih, (float*)d_out);
}
